// TransformerUnit_377957122743
// MI455X (gfx1250) — matrix operand order repaired and checked by reading, not compiled, not run on hardware
//
#include <hip/hip_runtime.h>
#include <math.h>

typedef __attribute__((ext_vector_type(16))) _Float16 v16h;
typedef __attribute__((ext_vector_type(8)))  float    v8f;
typedef __attribute__((ext_vector_type(2)))  _Float16 h2;

#define CCH 64
#define KS  5

// ---------------------------------------------------------------------------
// 0) tiny zero-fill (for the shared zero-bias buffer; avoids null checks)
// ---------------------------------------------------------------------------
__global__ void tu_zero_kernel(float* __restrict__ p, int n) {
  const int i = blockIdx.x * blockDim.x + threadIdx.x;
  if (i < n) p[i] = 0.0f;
}

// ---------------------------------------------------------------------------
// 1) DETR sine positional encoding -> pos[64, H*W] (channel-major f32)
// ---------------------------------------------------------------------------
__global__ void tu_pos_kernel(float* __restrict__ pos, int H, int W) {
  const int HW = H * W;
  const int idx = blockIdx.x * blockDim.x + threadIdx.x;
  if (idx >= CCH * HW) return;
  const int ch = idx / HW;
  const int p  = idx - ch * HW;
  const int y = p / W, x = p - (p / W) * W;
  const float TWO_PI = 6.283185307179586f;
  const int cc = ch & 31;
  float e = (ch < 32) ? ((float)(y + 1) * (TWO_PI / ((float)H + 1e-6f)))
                      : ((float)(x + 1) * (TWO_PI / ((float)W + 1e-6f)));
  const int j = cc >> 1;
  float t = powf(10000.0f, (float)j * (2.0f / 32.0f));
  float a = e / t;
  pos[idx] = (cc & 1) ? cosf(a) : sinf(a);
}

// ---------------------------------------------------------------------------
// 2) Weight pack: f32 [M,K] row-major -> f16 in WMMA A-fragment order.
//    Packed offset = (((mT*kSteps)+ks)*32 + lane)*16 + i, so each lane's
//    16-half fragment is one 32B contiguous chunk (contiguous K order:
//    lanes 0-15 hold K kk+0..15, lanes 16-31 hold kk+16..31, matching B).
// ---------------------------------------------------------------------------
__global__ void tu_pack_w_kernel(const float* __restrict__ W,
                                 _Float16* __restrict__ Wp, int M, int Kdim) {
  const int idx = blockIdx.x * blockDim.x + threadIdx.x;
  if (idx >= M * Kdim) return;
  const int kSteps = Kdim >> 5;
  const int i    = idx & 15;
  const int lane = (idx >> 4) & 31;
  const int rem  = idx >> 9;
  const int ks   = rem % kSteps;
  const int mT   = rem / kSteps;
  const int half = lane >> 4, l15 = lane & 15;
  const int mRow = (mT << 4) + l15;
  const int kOff = (half << 4) + i;
  Wp[idx] = (_Float16)W[mRow * Kdim + (ks << 5) + kOff];
}

// ---------------------------------------------------------------------------
// 3) Activation pack: f32 channel-major [n,64,HW] (+pos) -> f16 pixel-major
//    [n*HW, 64].  LDS 64x64 tile transpose: reads and writes both coalesced.
// ---------------------------------------------------------------------------
template<bool HASPOS>
__global__ void tu_pack_x_kernel(const float* __restrict__ X,
                                 const float* __restrict__ pos,
                                 _Float16* __restrict__ Xp,
                                 int HW) {                      // grid = n*HW/64
  __shared__ float tile[64][65];
  const int t = threadIdx.x;                 // 256 threads
  const int tilesPerBatch = HW >> 6;
  const int batch = blockIdx.x / tilesPerBatch;
  const int p0    = (blockIdx.x - batch * tilesPerBatch) << 6;
  const float* xb = X + (size_t)batch * CCH * HW;
  #pragma unroll
  for (int e = 0; e < 16; ++e) {
    const int idx = e * 256 + t;
    const int c = idx >> 6;                  // channel
    const int p = idx & 63;                  // pixel within tile
    float v = xb[c * HW + p0 + p];
    if (HASPOS) v += pos[c * HW + p0 + p];
    tile[p][c] = v;
  }
  __syncthreads();
  _Float16* yb = Xp + ((size_t)batch * HW + p0) * CCH;
  #pragma unroll
  for (int e = 0; e < 16; ++e) {
    const int idx = e * 256 + t;
    const int p = idx >> 6;
    const int c = idx & 63;
    yb[p * CCH + c] = (_Float16)tile[p][c];
  }
}

// ---------------------------------------------------------------------------
// 4) WMMA GEMM on pre-packed f16 operands.
//    A fragment: one 32B contiguous vector load per lane per k-step.
//    B fragment: 16 contiguous halves of this lane's column (pixel-major X).
//    Fully unrolled over KSTEPS -> KSTEPS back-to-back v_wmma per tile.
//    Epilogue: bias (always valid) + branchless leaky + optional f16 residual;
//    output either f16 pixel-major or f32 channel-major (final layer).
// ---------------------------------------------------------------------------
template<int KSTEPS, bool OUT_F16, bool HAS_RESID>
__global__ void tu_gemm_kernel(const _Float16* __restrict__ Wp,
                               const float*    __restrict__ bias,   // [M]
                               const _Float16* __restrict__ Xp,     // [rows, K]
                               const _Float16* __restrict__ resid,  // [rows, M]
                               float*          __restrict__ Yf,     // [n, M, HW]
                               _Float16*       __restrict__ Yh,     // [rows, M]
                               int M, int rows, int HW, float negSlope)
{
  const int Kdim = KSTEPS * 32;
  const int lane        = threadIdx.x & 31;
  const int wid         = blockIdx.x * (blockDim.x >> 5) + (threadIdx.x >> 5);
  const int mTiles = M >> 4;
  const int nTiles = rows >> 4;
  if (wid >= mTiles * nTiles) return;        // wave-uniform

  const int mT   = wid % mTiles;
  const int nT   = wid / mTiles;
  const int half = lane >> 4;
  const int l15  = lane & 15;
  const int col  = (nT << 4) + l15;          // global pixel for B / D

  const _Float16* xcol  = Xp + (size_t)col * Kdim;
  const _Float16* wtile = Wp + ((size_t)mT * KSTEPS * 32 + lane) * 16;

  v8f acc = {};
  #pragma unroll
  for (int ks = 0; ks < KSTEPS; ++ks) {
    v16h a = *(const v16h*)(wtile + (size_t)ks * 32 * 16);
    v16h b = *(const v16h*)(xcol + ks * 32 + (half << 4));
    acc = __builtin_amdgcn_wmma_f32_16x16x32_f16(false, a, false, b,
                                                 (short)0, acc, false, false);
  }

  const int mBase = (mT << 4) + (half << 3); // D: VGPR r -> M = mBase + r
  if (OUT_F16) {
    _Float16* yp = Yh + (size_t)col * M + mBase;
    const _Float16* rp = HAS_RESID ? (resid + (size_t)col * M + mBase) : nullptr;
    #pragma unroll
    for (int r = 0; r < 8; r += 2) {
      float v0 = acc[r]     + bias[mBase + r];
      float v1 = acc[r + 1] + bias[mBase + r + 1];
      v0 = fmaxf(v0, 0.0f) + negSlope * fminf(v0, 0.0f);
      v1 = fmaxf(v1, 0.0f) + negSlope * fminf(v1, 0.0f);
      if (HAS_RESID) { v0 += (float)rp[r]; v1 += (float)rp[r + 1]; }
      h2 st; st[0] = (_Float16)v0; st[1] = (_Float16)v1;
      *(h2*)(yp + r) = st;                   // 4B-aligned paired store
    }
  } else {
    const int batch = col / HW;
    const int p     = col - batch * HW;
    float* yb = Yf + ((size_t)batch * M + mBase) * HW + p;
    const _Float16* rp = HAS_RESID ? (resid + (size_t)col * M + mBase) : nullptr;
    #pragma unroll
    for (int r = 0; r < 8; ++r) {
      float vv = acc[r] + bias[mBase + r];
      vv = fmaxf(vv, 0.0f) + negSlope * fminf(vv, 0.0f);
      if (HAS_RESID) vv += (float)rp[r];
      yb[r * HW] = vv;
    }
  }
}

// ---------------------------------------------------------------------------
// 5) Flow-guided 5x5 attention, wave-per-pixel, online softmax, f16 I/O.
//    Pixel-major [n*HW, 64]: each bilinear tap is one dword (h2) per lane,
//    128B contiguous across the wave.  Head dot (dk=8 = 4 lanes x 2ch)
//    reduced with shfl_xor inside the wave32.
// ---------------------------------------------------------------------------
__global__ void tu_attn_kernel(const _Float16* __restrict__ Qp,
                               const _Float16* __restrict__ Kp,
                               const _Float16* __restrict__ Vp,
                               const float*    __restrict__ flow,  // [n,2,H,W]
                               _Float16*       __restrict__ outp,  // [n*HW, 64]
                               int nBatch, int H, int W)
{
  const int HW   = H * W;
  const int lane = threadIdx.x & 31;
  const int g    = blockIdx.x * (blockDim.x >> 5) + (threadIdx.x >> 5);
  if (g >= nBatch * HW) return;
  const int batch = g / HW;
  const int p     = g - batch * HW;
  const int y = p / W, x = p - (p / W) * W;

  const float fx = flow[(batch * 2 + 0) * HW + p];
  const float fy = flow[(batch * 2 + 1) * HW + p];

  const int c0 = lane * 2;
  const h2 qh = *(const h2*)(Qp + (size_t)g * CCH + c0);
  const float q0 = (float)qh[0], q1 = (float)qh[1];

  float m = -1e30f, l = 0.0f, a0 = 0.0f, a1 = 0.0f;
  const float scale = 0.3535533905932738f;   // 1/sqrt(dk=8)

  for (int ky = 0; ky < KS; ++ky) {
    for (int kx = 0; kx < KS; ++kx) {
      const float xs = (float)x + (float)(kx - 2) + fx;
      const float ys = (float)y + (float)(ky - 2) + fy;
      const float x0f = floorf(xs), y0f = floorf(ys);
      const float wx = xs - x0f, wy = ys - y0f;
      const int x0 = (int)x0f, y0 = (int)y0f;

      float k0 = 0.f, k1 = 0.f, v0 = 0.f, v1 = 0.f;
      #pragma unroll
      for (int t = 0; t < 4; ++t) {
        const int yy = y0 + (t >> 1);
        const int xx = x0 + (t & 1);
        const float wt = ((t >> 1) ? wy : 1.0f - wy) * ((t & 1) ? wx : 1.0f - wx);
        if (xx >= 0 && xx < W && yy >= 0 && yy < H) {  // wave-uniform branch
          const size_t base = ((size_t)batch * HW + yy * W + xx) * CCH + c0;
          const h2 kh = *(const h2*)(Kp + base);
          const h2 vh = *(const h2*)(Vp + base);
          k0 += wt * (float)kh[0];  k1 += wt * (float)kh[1];
          v0 += wt * (float)vh[0];  v1 += wt * (float)vh[1];
        }
      }

      float s = q0 * k0 + q1 * k1;           // partial: 2 of 8 channels
      s += __shfl_xor(s, 1, 32);             // reduce dk across 4-lane head
      s += __shfl_xor(s, 2, 32);
      s *= scale;

      const float mn = fmaxf(m, s);          // online softmax
      const float f  = __expf(m - mn);
      const float e  = __expf(s - mn);
      l  = l * f + e;
      a0 = a0 * f + e * v0;
      a1 = a1 * f + e * v1;
      m  = mn;
    }
  }

  const float inv = 1.0f / l;
  h2 st; st[0] = (_Float16)(a0 * inv); st[1] = (_Float16)(a1 * inv);
  *(h2*)(outp + (size_t)g * CCH + c0) = st;
}

// ---------------------------------------------------------------------------
// 6) GroupNorm(1,C) == LayerNorm over (C,H,W), in place, channel-major f32.
// ---------------------------------------------------------------------------
__global__ void tu_groupnorm_kernel(float* __restrict__ data,
                                    const float* __restrict__ gw,
                                    const float* __restrict__ gb,
                                    int C, int HW)
{
  const int total = C * HW;
  float* d = data + (size_t)blockIdx.x * total;

  float s = 0.0f, s2 = 0.0f;
  for (int i = threadIdx.x; i < total; i += blockDim.x) {
    const float v = d[i];
    s += v; s2 += v * v;
  }
  for (int o = 16; o > 0; o >>= 1) {
    s  += __shfl_xor(s,  o, 32);
    s2 += __shfl_xor(s2, o, 32);
  }
  __shared__ float rs[32], rs2[32];
  const int wave = threadIdx.x >> 5, lane = threadIdx.x & 31;
  if (lane == 0) { rs[wave] = s; rs2[wave] = s2; }
  __syncthreads();
  const int nw = blockDim.x >> 5;
  if (wave == 0) {
    s  = (lane < nw) ? rs[lane]  : 0.0f;
    s2 = (lane < nw) ? rs2[lane] : 0.0f;
    for (int o = 16; o > 0; o >>= 1) {
      s  += __shfl_xor(s,  o, 32);
      s2 += __shfl_xor(s2, o, 32);
    }
    if (lane == 0) { rs[0] = s; rs2[0] = s2; }
  }
  __syncthreads();
  const float mu  = rs[0] / (float)total;
  const float var = rs2[0] / (float)total - mu * mu;
  const float inv = rsqrtf(var + 1e-5f);
  for (int i = threadIdx.x; i < total; i += blockDim.x) {
    const int c = i / HW;
    d[i] = (d[i] - mu) * inv * gw[c] + gb[c];
  }
}

// ---------------------------------------------------------------------------
extern "C" void kernel_launch(void* const* d_in, const int* in_sizes, int n_in,
                              void* d_out, int out_size, void* d_ws, size_t ws_size,
                              hipStream_t stream)
{
  const float* q      = (const float*)d_in[0];
  const float* k      = (const float*)d_in[1];
  const float* v      = (const float*)d_in[2];
  const float* flow   = (const float*)d_in[3];
  const float* w_qs   = (const float*)d_in[4];
  const float* w_ks   = (const float*)d_in[5];
  const float* w_vs   = (const float*)d_in[6];
  const float* w_fc   = (const float*)d_in[7];
  const float* mlp_w1 = (const float*)d_in[8];
  const float* mlp_b1 = (const float*)d_in[9];
  const float* mlp_w2 = (const float*)d_in[10];
  const float* mlp_b2 = (const float*)d_in[11];
  const float* norm_w = (const float*)d_in[12];
  const float* norm_b = (const float*)d_in[13];

  const int n = 2, C = 64, H = 96, W = 96, HW = H * W;
  const int rows = n * HW;                       // 18432 pixels

  // ---- workspace carve (256B aligned chunks) ----
  uintptr_t cur = (uintptr_t)d_ws;
  auto carve = [&](size_t bytes) -> void* {
    uintptr_t p = (cur + 255) & ~(uintptr_t)255;
    cur = p + bytes;
    return (void*)p;
  };
  float*     pos  = (float*)    carve((size_t)C * HW * 4);
  float*     zb   = (float*)    carve(128 * 4);            // zero bias
  _Float16*  Wq_p = (_Float16*) carve((size_t)64 * 64 * 2);
  _Float16*  Wk_p = (_Float16*) carve((size_t)64 * 64 * 2);
  _Float16*  Wv_p = (_Float16*) carve((size_t)64 * 64 * 2);
  _Float16*  Wf_p = (_Float16*) carve((size_t)64 * 64 * 2);
  _Float16*  W1_p = (_Float16*) carve((size_t)128 * 64 * 2);
  _Float16*  W2_p = (_Float16*) carve((size_t)64 * 128 * 2);
  _Float16*  Xq   = (_Float16*) carve((size_t)rows * 64 * 2);
  _Float16*  Xk   = (_Float16*) carve((size_t)rows * 64 * 2);
  _Float16*  Xv   = (_Float16*) carve((size_t)rows * 64 * 2);
  _Float16*  Qh   = (_Float16*) carve((size_t)rows * 64 * 2);
  _Float16*  Kh   = (_Float16*) carve((size_t)rows * 64 * 2);
  _Float16*  Vh   = (_Float16*) carve((size_t)rows * 64 * 2);
  _Float16*  att  = (_Float16*) carve((size_t)rows * 64 * 2);
  _Float16*  fc   = (_Float16*) carve((size_t)rows * 64 * 2);
  _Float16*  hdn  = (_Float16*) carve((size_t)rows * 128 * 2);

  // ---- prep: zero bias, pos-enc, pack weights, pack activations ----
  tu_zero_kernel<<<1, 128, 0, stream>>>(zb, 128);
  tu_pos_kernel<<<(C * HW + 255) / 256, 256, 0, stream>>>(pos, H, W);

  tu_pack_w_kernel<<<(64 * 64)  / 256, 256, 0, stream>>>(w_qs,   Wq_p, 64, 64);
  tu_pack_w_kernel<<<(64 * 64)  / 256, 256, 0, stream>>>(w_ks,   Wk_p, 64, 64);
  tu_pack_w_kernel<<<(64 * 64)  / 256, 256, 0, stream>>>(w_vs,   Wv_p, 64, 64);
  tu_pack_w_kernel<<<(64 * 64)  / 256, 256, 0, stream>>>(w_fc,   Wf_p, 64, 64);
  tu_pack_w_kernel<<<(128 * 64) / 256, 256, 0, stream>>>(mlp_w1, W1_p, 128, 64);
  tu_pack_w_kernel<<<(64 * 128) / 256, 256, 0, stream>>>(mlp_w2, W2_p, 64, 128);

  const int xBlocks = rows / 64;                 // 288 LDS-transpose tiles
  tu_pack_x_kernel<true ><<<xBlocks, 256, 0, stream>>>(q, pos, Xq, HW);
  tu_pack_x_kernel<true ><<<xBlocks, 256, 0, stream>>>(k, pos, Xk, HW);
  tu_pack_x_kernel<false><<<xBlocks, 256, 0, stream>>>(v, pos, Xv, HW);

  auto gemmBlocks = [](int M, int r) {
    const int waves = (M / 16) * (r / 16);
    return (waves + 7) / 8;                      // 256 threads = 8 waves
  };

  // ---- projections (f16 pixel-major out) ----
  tu_gemm_kernel<2, true, false><<<gemmBlocks(64, rows), 256, 0, stream>>>(
      Wq_p, zb, Xq, nullptr, nullptr, Qh, 64, rows, HW, 1.0f);
  tu_gemm_kernel<2, true, false><<<gemmBlocks(64, rows), 256, 0, stream>>>(
      Wk_p, zb, Xk, nullptr, nullptr, Kh, 64, rows, HW, 1.0f);
  tu_gemm_kernel<2, true, false><<<gemmBlocks(64, rows), 256, 0, stream>>>(
      Wv_p, zb, Xv, nullptr, nullptr, Vh, 64, rows, HW, 1.0f);

  // ---- flow-guided attention (wave-per-pixel) ----
  tu_attn_kernel<<<(rows + 7) / 8, 256, 0, stream>>>(Qh, Kh, Vh, flow, att, n, H, W);

  // ---- fc projection ----
  tu_gemm_kernel<2, true, false><<<gemmBlocks(64, rows), 256, 0, stream>>>(
      Wf_p, zb, att, nullptr, nullptr, fc, 64, rows, HW, 1.0f);

  // ---- MLP layer 1: LeakyReLU(W1 @ fc + b1), M=128 ----
  tu_gemm_kernel<2, true, false><<<gemmBlocks(128, rows), 256, 0, stream>>>(
      W1_p, mlp_b1, fc, nullptr, nullptr, hdn, 128, rows, HW, 0.2f);

  // ---- MLP layer 2 + residual -> d_out (channel-major f32), K=128 ----
  tu_gemm_kernel<4, false, true><<<gemmBlocks(64, rows), 256, 0, stream>>>(
      W2_p, mlp_b2, hdn, fc, (float*)d_out, nullptr, 64, rows, HW, 1.0f);

  // ---- GroupNorm(1, C) in place ----
  tu_groupnorm_kernel<<<n, 256, 0, stream>>>((float*)d_out, norm_w, norm_b, C, HW);
}
